// SFIModel_gating_8478265442836
// MI455X (gfx1250) — matrix operand order repaired and checked by reading, not compiled, not run on hardware
//
#include <hip/hip_runtime.h>
#include <math.h>

// ---------------------------------------------------------------------------
// Problem constants
// ---------------------------------------------------------------------------
#define Bx    32
#define CDD   5
#define HIS   50
#define Sx    20
#define Ex    300
#define Fx    150
#define Kx    30
#define Lx    3
// Derived
#define NSEQ  (Bx * (CDD + HIS))          // 1760 sequences (first 160 = candidates)
#define NROW  (NSEQ * Sx)                 // 35200 token rows
#define FP    160                         // F padded to multiple of 32
#define K1P   928                         // 3*300=900 padded to mult of 32
#define K23P  480                         // 3*160
#define NUNIT (Bx * CDD * Kx * Lx)        // 14400 fusion tiles

typedef __attribute__((ext_vector_type(16))) _Float16 v16h;
typedef __attribute__((ext_vector_type(8)))  float    v8f;

union HF16 { v16h v; _Float16 h[16]; };

__device__ __forceinline__ int amap_k(int vv, int hi) {
  // 16-bit A-matrix 16x32 layout (ISA 7.12.2): lanes 0-15 -> K 0..7 / 16..23,
  // lanes 16-31 -> K 8..15 / 24..31, packed as consecutive pairs per VGPR.
  return ((vv < 4) ? (2 * vv) : (16 + 2 * (vv - 4))) + 8 * hi;
}

__device__ __forceinline__ v8f wmma32(v16h a, v16h b, v8f c) {
  return __builtin_amdgcn_wmma_f32_16x16x32_f16(false, a, false, b, (short)0, c,
                                                false, false);
}

__device__ __forceinline__ float wave_sum(float v) {
#pragma unroll
  for (int o = 16; o; o >>= 1) v += __shfl_xor(v, o, 32);
  return v;
}

// gfx1250 async global->LDS copy, one b128 per active lane (ISA 10.x / 15.18.3)
__device__ __forceinline__ void async_copy_b128(unsigned lds_off,
                                                const void* gaddr) {
  asm volatile("global_load_async_to_lds_b128 %0, %1, off"
               :
               : "v"(lds_off), "v"(gaddr)
               : "memory");
}

__device__ __forceinline__ void wait_asynccnt0() {
  asm volatile("s_wait_asynccnt 0x0" ::: "memory");
}

// A fragment from row-major (M x K) f16 matrix, lda in elements
__device__ __forceinline__ v16h load_a(const _Float16* A, int lda, int mbase,
                                       int kb, int lane) {
  HF16 r;
  int m  = mbase + (lane & 15);
  int hi = (lane >> 4) & 1;
  const _Float16* row = A + (long)m * lda + kb;
#pragma unroll
  for (int vv = 0; vv < 8; ++vv) {
    int k           = amap_k(vv, hi);
    r.h[2 * vv]     = row[k];
    r.h[2 * vv + 1] = row[k + 1];
  }
  return r.v;
}

// B fragment (K x N logical) sourced from row-major (N x K) operand ->
// consecutive-K pairs are contiguous.  ldb in elements.
__device__ __forceinline__ v16h load_b(const _Float16* Bt, int ldb, int nbase,
                                       int kb, int lane) {
  HF16 r;
  int n  = nbase + (lane & 15);
  int hi = (lane >> 4) & 1;
  const _Float16* row = Bt + (long)n * ldb + kb;
#pragma unroll
  for (int vv = 0; vv < 8; ++vv) {
    int k           = amap_k(vv, hi);
    r.h[2 * vv]     = row[k];
    r.h[2 * vv + 1] = row[k + 1];
  }
  return r.v;
}

// Guarded + ReLU'd variants for the 20-row fusion tiles
__device__ __forceinline__ v16h load_a_relu20(const _Float16* base, int mbase,
                                              int kb, int lane) {
  HF16 r;
  int m  = mbase + (lane & 15);
  int hi = (lane >> 4) & 1;
  if (m >= Sx) {
#pragma unroll
    for (int i = 0; i < 16; ++i) r.h[i] = (_Float16)0;
    return r.v;
  }
  const _Float16* row = base + (long)m * FP + kb;
#pragma unroll
  for (int vv = 0; vv < 8; ++vv) {
    int k       = amap_k(vv, hi);
    _Float16 x0 = row[k], x1 = row[k + 1];
    r.h[2 * vv]     = x0 > (_Float16)0 ? x0 : (_Float16)0;
    r.h[2 * vv + 1] = x1 > (_Float16)0 ? x1 : (_Float16)0;
  }
  return r.v;
}

__device__ __forceinline__ v16h load_b_relu20(const _Float16* base, int nbase,
                                              int kb, int lane) {
  HF16 r;
  int n  = nbase + (lane & 15);
  int hi = (lane >> 4) & 1;
  if (n >= Sx) {
#pragma unroll
    for (int i = 0; i < 16; ++i) r.h[i] = (_Float16)0;
    return r.v;
  }
  const _Float16* row = base + (long)n * FP + kb;
#pragma unroll
  for (int vv = 0; vv < 8; ++vv) {
    int k       = amap_k(vv, hi);
    _Float16 x0 = row[k], x1 = row[k + 1];
    r.h[2 * vv]     = x0 > (_Float16)0 ? x0 : (_Float16)0;
    r.h[2 * vv + 1] = x1 > (_Float16)0 ? x1 : (_Float16)0;
  }
  return r.v;
}

// ---------------------------------------------------------------------------
// K0: pack conv1d weights (F, Cin, 3) f32 -> f16 as Bt[f][k], k = dt*segPad+c
// ---------------------------------------------------------------------------
__global__ __launch_bounds__(256) void pack_w_kernel(const float* w,
                                                     _Float16* Bt, int seg,
                                                     int segPad, int Kp) {
  long total = (long)FP * Kp;
  for (long i = (long)blockIdx.x * blockDim.x + threadIdx.x; i < total;
       i += (long)gridDim.x * blockDim.x) {
    int f = (int)(i / Kp), k = (int)(i % Kp);
    float v = 0.f;
    if (f < Fx && k < 3 * segPad) {
      int dt = k / segPad, c = k % segPad;
      if (c < seg) v = w[(long)f * seg * 3 + (long)c * 3 + dt];
    }
    Bt[i] = (_Float16)v;
  }
}

// ---------------------------------------------------------------------------
// K1: embedding gather + im2col for conv layer 1 (dilation 1, pad 1)
//     A1[row=seq*20+t][k = dt*300 + e] f16, K padded 900->928
// ---------------------------------------------------------------------------
__global__ __launch_bounds__(128) void im2col1_kernel(const int* cand,
                                                      const int* clk,
                                                      const float* emb,
                                                      _Float16* A1) {
  int r = blockIdx.x;  // 0..NROW-1
  int seq = r / Sx, t = r % Sx;
  const int* ids = (seq < Bx * CDD) ? (cand + seq * Sx)
                                    : (clk + (seq - Bx * CDD) * Sx);
  int id[3];
#pragma unroll
  for (int dt = 0; dt < 3; ++dt) {
    int tok = t + dt - 1;
    id[dt]  = (tok >= 0 && tok < Sx) ? ids[tok] : -1;
  }
  _Float16* row = A1 + (long)r * K1P;
  for (int k = threadIdx.x; k < K1P; k += 128) {
    float v = 0.f;
    if (k < 3 * Ex) {
      int dt = k / Ex, e = k - dt * Ex;
      if (id[dt] >= 0) v = emb[(long)id[dt] * Ex + e];
    }
    row[k] = (_Float16)v;
  }
}

// ---------------------------------------------------------------------------
// K2: WMMA GEMM for conv layer 1 with async-LDS A staging.
//     One block per 16-row M tile; the A tile (16x928 f16 = 29.7 KB) is
//     DMA'd global->LDS once via global_load_async_to_lds_b128 and consumed
//     by all 10 N-tile waves.  B stays in global (290 KB, L2-resident).
// ---------------------------------------------------------------------------
__global__ __launch_bounds__(320) void gemm1_lds_kernel(const _Float16* A,
                                                        const _Float16* Bt,
                                                        const float* bias,
                                                        float* C) {
  __shared__ __align__(16) _Float16 sA[16 * K1P];  // 29696 bytes
  const int tm  = blockIdx.x;  // 0..2199
  const int tid = threadIdx.x;

  // async stage: contiguous 29696-byte tile, 16 B per lane per instruction
  {
    const char* src = (const char*)(A + (long)tm * 16 * K1P);
    unsigned    dst = (unsigned)(uintptr_t)(&sA[0]);
    const int   BYTES = 16 * K1P * 2;
    for (int off = tid * 16; off < BYTES; off += 320 * 16)
      async_copy_b128(dst + off, src + off);
  }
  wait_asynccnt0();   // own async ops done
  __syncthreads();    // everyone's async ops done

  const int w = tid >> 5;        // wave = N tile, 0..9
  const int lane = tid & 31;
  v8f acc = {};
#pragma unroll 1
  for (int kb = 0; kb < K1P; kb += 32) {
    // A fragment from LDS (ds_load), B fragment from global
    HF16 ra;
    {
      int m  = lane & 15;
      int hi = (lane >> 4) & 1;
      const _Float16* row = sA + m * K1P + kb;
#pragma unroll
      for (int vv = 0; vv < 8; ++vv) {
        int k            = amap_k(vv, hi);
        ra.h[2 * vv]     = row[k];
        ra.h[2 * vv + 1] = row[k + 1];
      }
    }
    v16h b = load_b(Bt, K1P, w * 16, kb, lane);
    acc    = wmma32(ra.v, b, acc);
  }
  int n    = w * 16 + (lane & 15);
  int hi   = (lane >> 4) & 1;
  float bv = (n < Fx) ? bias[n] : 0.f;
#pragma unroll
  for (int r = 0; r < 8; ++r) {
    int m = tm * 16 + r + 8 * hi;
    C[(long)m * FP + n] = acc[r] + bv;
  }
}

// ---------------------------------------------------------------------------
// K2': WMMA GEMM for dilated conv2/3: A gathered on the fly from padded d
//      (NROW x 160 f16).  K=480 = 3 segments of 160 (dt = -dil, 0, +dil);
//      160 % 32 == 0 so each K chunk lives in exactly one segment.
// ---------------------------------------------------------------------------
__global__ __launch_bounds__(256) void gemm_dilated_kernel(const _Float16* D,
                                                           const _Float16* Bt,
                                                           const float* bias,
                                                           float* C, int dil) {
  const int Ntiles = FP / 16;
  const int Mtiles = NROW / 16;
  int wid  = blockIdx.x * (blockDim.x >> 5) + (threadIdx.x >> 5);
  int lane = threadIdx.x & 31;
  if (wid >= Mtiles * Ntiles) return;
  int tm = wid / Ntiles, tn = wid % Ntiles;
  int m   = tm * 16 + (lane & 15);
  int hi  = (lane >> 4) & 1;
  int seq = m / Sx, t = m % Sx;
  v8f acc = {};
#pragma unroll 1
  for (int kb = 0; kb < K23P; kb += 32) {
    int seg = kb / FP;
    int cb  = kb - seg * FP;
    int ts  = t + (seg - 1) * dil;
    HF16 ra;
    if (ts < 0 || ts >= Sx) {
#pragma unroll
      for (int i = 0; i < 16; ++i) ra.h[i] = (_Float16)0;
    } else {
      const _Float16* row = D + ((long)(seq * Sx + ts)) * FP + cb;
#pragma unroll
      for (int vv = 0; vv < 8; ++vv) {
        int k            = amap_k(vv, hi);
        ra.h[2 * vv]     = row[k];
        ra.h[2 * vv + 1] = row[k + 1];
      }
    }
    v16h b = load_b(Bt, K23P, tn * 16, kb, lane);
    acc    = wmma32(ra.v, b, acc);
  }
  int n    = tn * 16 + (lane & 15);
  float bv = (n < Fx) ? bias[n] : 0.f;
#pragma unroll
  for (int r = 0; r < 8; ++r) {
    int mm = tm * 16 + r + 8 * hi;
    C[(long)mm * FP + n] = acc[r] + bv;
  }
}

// ---------------------------------------------------------------------------
// K3: row-wise LayerNorm over F=150, write f16 padded to 160 (zeros)
// ---------------------------------------------------------------------------
__global__ __launch_bounds__(256) void ln_kernel(const float* pre,
                                                 const float* gw,
                                                 const float* gb,
                                                 _Float16* Dst) {
  int row = blockIdx.x * (blockDim.x >> 5) + (threadIdx.x >> 5);
  if (row >= NROW) return;
  int lane       = threadIdx.x & 31;
  const float* x = pre + (long)row * FP;
  float s = 0.f, s2 = 0.f;
  for (int k = lane; k < Fx; k += 32) {
    float v = x[k];
    s += v;
    s2 += v * v;
  }
  s  = wave_sum(s);
  s2 = wave_sum(s2);
  float m   = s * (1.f / Fx);
  float var = s2 * (1.f / Fx) - m * m;
  float inv = rsqrtf(var + 1e-5f);
  _Float16* dr = Dst + (long)row * FP;
  for (int k = lane; k < FP; k += 32) {
    float v = (k < Fx) ? ((x[k] - m) * inv * gw[k] + gb[k]) : 0.f;
    dr[k]   = (_Float16)v;
  }
}

// ---------------------------------------------------------------------------
// K4: level attention + word attention -> per-sequence representation (f32)
// ---------------------------------------------------------------------------
__global__ __launch_bounds__(256) void repr_kernel(const _Float16* D1,
                                                   const _Float16* D2,
                                                   const _Float16* D3,
                                                   const float* qlev,
                                                   const float* qwrd,
                                                   float* reprs) {
  int seq  = blockIdx.x;
  int tid  = threadIdx.x, lane = tid & 31, w = tid >> 5;
  __shared__ float llog[Sx][4];
  __shared__ float lws[Sx][4];
  __shared__ float wf[Sx][152];
  __shared__ float wlog[Sx];
  __shared__ float ww[Sx];
  const _Float16* Dl[3] = {D1 + (long)seq * Sx * FP, D2 + (long)seq * Sx * FP,
                           D3 + (long)seq * Sx * FP};
  const float inv_se = 0.05773502691896258f;  // 1/sqrt(300)

  for (int p = w; p < Sx * Lx; p += 8) {
    int t = p / Lx, l = p % Lx;
    const _Float16* d = Dl[l] + t * FP;
    float s = 0.f;
    for (int f = lane; f < Fx; f += 32) {
      float v = (float)d[f];
      v       = v > 0.f ? v : 0.f;
      s += v * qlev[f];
    }
    s = wave_sum(s);
    if (lane == 0) llog[t][l] = s * inv_se;
  }
  __syncthreads();
  if (tid < Sx) {
    int t     = tid;
    float a   = llog[t][0], b = llog[t][1], c = llog[t][2];
    float mx  = fmaxf(a, fmaxf(b, c));
    float ea  = expf(a - mx), eb = expf(b - mx), ec = expf(c - mx);
    float sum = ea + eb + ec;
    lws[t][0] = ea / sum;
    lws[t][1] = eb / sum;
    lws[t][2] = ec / sum;
  }
  __syncthreads();
  for (int p = tid; p < Sx * Fx; p += 256) {
    int t = p / Fx, f = p % Fx;
    float acc = 0.f;
#pragma unroll
    for (int l = 0; l < Lx; ++l) {
      float v = (float)Dl[l][t * FP + f];
      v       = v > 0.f ? v : 0.f;
      acc += lws[t][l] * v;
    }
    wf[t][f] = acc;
  }
  __syncthreads();
  for (int t = w; t < Sx; t += 8) {
    float s = 0.f;
    for (int f = lane; f < Fx; f += 32) s += wf[t][f] * qwrd[f];
    s = wave_sum(s);
    if (lane == 0) wlog[t] = s * inv_se;
  }
  __syncthreads();
  if (w == 0) {
    float v  = (lane < Sx) ? wlog[lane] : -INFINITY;
    float mx = v;
#pragma unroll
    for (int o = 16; o; o >>= 1) mx = fmaxf(mx, __shfl_xor(mx, o, 32));
    float e = (lane < Sx) ? expf(v - mx) : 0.f;
    float s = wave_sum(e);
    if (lane < Sx) ww[lane] = e / s;
  }
  __syncthreads();
  for (int f = tid; f < FP; f += 256) {
    float acc = 0.f;
    if (f < Fx)
      for (int t = 0; t < Sx; ++t) acc += ww[t] * wf[t][f];
    reprs[(long)seq * FP + f] = acc;
  }
}

// ---------------------------------------------------------------------------
// K5: candidate-history scores + top-K index selection (softmax is monotone,
//     only the indices of top-K are consumed downstream)
// ---------------------------------------------------------------------------
__global__ __launch_bounds__(256) void topk_kernel(const float* reprs,
                                                   const unsigned char* mask,
                                                   int* idx) {
  int bc = blockIdx.x;  // 0..159
  int b  = bc / CDD;
  __shared__ float sc[64];
  int tid = threadIdx.x, lane = tid & 31, w = tid >> 5;
  const float* cr = reprs + (long)bc * FP;
  for (int h = w; h < HIS; h += 8) {
    const float* hr = reprs + (long)(Bx * CDD + b * HIS + h) * FP;
    float s = 0.f;
    for (int f = lane; f < Fx; f += 32) s += cr[f] * hr[f];
    s = wave_sum(s);
    if (lane == 0) sc[h] = mask[b * HIS + h] ? -INFINITY : s;
  }
  __syncthreads();
  if (w == 0) {
    float v0 = sc[lane];
    float v1 = (lane + 32 < HIS) ? sc[lane + 32] : -INFINITY;
    for (int k = 0; k < Kx; ++k) {
      float best = v0;
      int bi     = lane;
      if (v1 > best) { best = v1; bi = lane + 32; }
#pragma unroll
      for (int o = 16; o; o >>= 1) {
        float ov = __shfl_xor(best, o, 32);
        int   oi = __shfl_xor(bi, o, 32);
        if (ov > best || (ov == best && oi < bi)) { best = ov; bi = oi; }
      }
      if (lane == 0) idx[bc * Kx + k] = bi;
      if (bi == lane) v0 = -INFINITY;
      if (bi == lane + 32) v1 = -INFINITY;
    }
  }
}

// ---------------------------------------------------------------------------
// K6: fusion einsum -> (160, L, K, 20, 20) f32.  One wave per (bc,k,l),
//     2x2 WMMA tiles (20 padded to 32), K = 160 (150 + zero pad), ReLU in the
//     fragment loads.
// ---------------------------------------------------------------------------
__global__ __launch_bounds__(256) void fusion_kernel(const _Float16* D1,
                                                     const _Float16* D2,
                                                     const _Float16* D3,
                                                     const int* idx,
                                                     float* Fz) {
  int unit = blockIdx.x * (blockDim.x >> 5) + (threadIdx.x >> 5);
  if (unit >= NUNIT) return;
  int lane = threadIdx.x & 31;
  int bc  = unit / (Kx * Lx);
  int rem = unit % (Kx * Lx);
  int k = rem / Lx, l = rem % Lx;
  int b = bc / CDD;
  int h = idx[bc * Kx + k];
  const _Float16* Dl = (l == 0) ? D1 : (l == 1) ? D2 : D3;
  const _Float16* Ar = Dl + (long)bc * Sx * FP;
  const _Float16* Br = Dl + (long)(Bx * CDD + b * HIS + h) * Sx * FP;
  v8f acc[2][2] = {{{}, {}}, {{}, {}}};
#pragma unroll
  for (int kb = 0; kb < FP; kb += 32) {
    v16h a0 = load_a_relu20(Ar, 0, kb, lane);
    v16h a1 = load_a_relu20(Ar, 16, kb, lane);
    v16h b0 = load_b_relu20(Br, 0, kb, lane);
    v16h b1 = load_b_relu20(Br, 16, kb, lane);
    acc[0][0] = wmma32(a0, b0, acc[0][0]);
    acc[0][1] = wmma32(a0, b1, acc[0][1]);
    acc[1][0] = wmma32(a1, b0, acc[1][0]);
    acc[1][1] = wmma32(a1, b1, acc[1][1]);
  }
  const float scale = 0.08164965809277261f;  // 1/sqrt(150)
  float* out = Fz + ((((long)bc * Lx + l) * Kx + k) * Sx * Sx);
  int hi = (lane >> 4) & 1;
#pragma unroll
  for (int tm = 0; tm < 2; ++tm) {
#pragma unroll
    for (int tn = 0; tn < 2; ++tn) {
      int n = tn * 16 + (lane & 15);
      if (n >= Sx) continue;
#pragma unroll
      for (int r = 0; r < 8; ++r) {
        int m = tm * 16 + r + 8 * hi;
        if (m < Sx) out[m * Sx + n] = acc[tm][tn][r] * scale;
      }
    }
  }
}

// ---------------------------------------------------------------------------
// K7: conv3d (3->32, 3x3x3, same pad) + ReLU + maxpool3 fused.
// ---------------------------------------------------------------------------
__global__ __launch_bounds__(256) void conv3d1_kernel(const float* X,
                                                      const float* W,
                                                      const float* bias,
                                                      float* P) {
  __shared__ float sw[32 * 81 + 32];
  for (int i = threadIdx.x; i < 32 * 81; i += 256) sw[i] = W[i];
  if (threadIdx.x < 32) sw[32 * 81 + threadIdx.x] = bias[threadIdx.x];
  __syncthreads();
  long gid = (long)blockIdx.x * 256 + threadIdx.x;  // 160*32*360 exact
  int px = (int)(gid % 6);
  int py = (int)((gid / 6) % 6);
  int pz = (int)((gid / 36) % 10);
  int oc = (int)((gid / 360) % 32);
  int n  = (int)(gid / (360 * 32));
  const float* xin = X + (long)n * Lx * Kx * Sx * Sx;
  const float* w   = sw + oc * 81;
  float bv         = sw[32 * 81 + oc];
  float best       = -INFINITY;
#pragma unroll 1
  for (int dz = 0; dz < 3; ++dz)
#pragma unroll 1
    for (int dy = 0; dy < 3; ++dy)
#pragma unroll 1
      for (int dx = 0; dx < 3; ++dx) {
        int z = pz * 3 + dz, y = py * 3 + dy, x = px * 3 + dx;
        float acc = bv;
#pragma unroll 1
        for (int ci = 0; ci < 3; ++ci)
#pragma unroll 1
          for (int kz = 0; kz < 3; ++kz)
#pragma unroll 1
            for (int ky = 0; ky < 3; ++ky)
#pragma unroll
              for (int kx = 0; kx < 3; ++kx) {
                int zz = z + kz - 1, yy = y + ky - 1, xx = x + kx - 1;
                if (zz < 0 || zz >= Kx || yy < 0 || yy >= Sx || xx < 0 ||
                    xx >= Sx)
                  continue;
                acc += xin[((ci * Kx + zz) * Sx + yy) * Sx + xx] *
                       w[((ci * 3 + kz) * 3 + ky) * 3 + kx];
              }
        acc  = acc > 0.f ? acc : 0.f;
        best = fmaxf(best, acc);
      }
  P[gid] = best;
}

// ---------------------------------------------------------------------------
// K8: conv3d (32->16) + ReLU + maxpool3 fused on (10,6,6) -> (3,2,2)
// ---------------------------------------------------------------------------
__global__ __launch_bounds__(256) void conv3d2_kernel(const float* X,
                                                      const float* W,
                                                      const float* bias,
                                                      float* P) {
  __shared__ float sw[16 * 864 + 16];
  for (int i = threadIdx.x; i < 16 * 864; i += 256) sw[i] = W[i];
  if (threadIdx.x < 16) sw[16 * 864 + threadIdx.x] = bias[threadIdx.x];
  __syncthreads();
  long gid = (long)blockIdx.x * 256 + threadIdx.x;  // 160*16*12 exact
  int px = (int)(gid % 2);
  int py = (int)((gid / 2) % 2);
  int pz = (int)((gid / 4) % 3);
  int oc = (int)((gid / 12) % 16);
  int n  = (int)(gid / 192);
  const float* xin = X + (long)n * 32 * 360;
  const float* w   = sw + oc * 864;
  float bv         = sw[16 * 864 + oc];
  float best       = -INFINITY;
#pragma unroll 1
  for (int dz = 0; dz < 3; ++dz)
#pragma unroll 1
    for (int dy = 0; dy < 3; ++dy)
#pragma unroll 1
      for (int dx = 0; dx < 3; ++dx) {
        int z = pz * 3 + dz, y = py * 3 + dy, x = px * 3 + dx;
        float acc = bv;
#pragma unroll 1
        for (int ci = 0; ci < 32; ++ci)
#pragma unroll 1
          for (int kz = 0; kz < 3; ++kz)
#pragma unroll 1
            for (int ky = 0; ky < 3; ++ky)
#pragma unroll
              for (int kx = 0; kx < 3; ++kx) {
                int zz = z + kz - 1, yy = y + ky - 1, xx = x + kx - 1;
                if (zz < 0 || zz >= 10 || yy < 0 || yy >= 6 || xx < 0 ||
                    xx >= 6)
                  continue;
                acc += xin[((ci * 10 + zz) * 6 + yy) * 6 + xx] *
                       w[((ci * 3 + kz) * 3 + ky) * 3 + kx];
              }
        acc  = acc > 0.f ? acc : 0.f;
        best = fmaxf(best, acc);
      }
  P[gid] = best;
}

// ---------------------------------------------------------------------------
// K9: final linear (192-d dot) + log_softmax over the 5 candidates
// ---------------------------------------------------------------------------
__global__ __launch_bounds__(32) void final_kernel(const float* feat,
                                                   const float* ltr_w,
                                                   const float* ltr_b,
                                                   float* out) {
  int b    = blockIdx.x;
  int lane = threadIdx.x;
  float sc[CDD];
#pragma unroll
  for (int c = 0; c < CDD; ++c) {
    const float* f = feat + ((long)b * CDD + c) * 192;
    float s = 0.f;
    for (int d = lane; d < 192; d += 32) s += f[d] * ltr_w[d];
    sc[c] = wave_sum(s) + ltr_b[0];
  }
  float mx = sc[0];
#pragma unroll
  for (int c = 1; c < CDD; ++c) mx = fmaxf(mx, sc[c]);
  float se = 0.f;
#pragma unroll
  for (int c = 0; c < CDD; ++c) se += expf(sc[c] - mx);
  float lse = mx + logf(se);
  if (lane < CDD) out[b * CDD + lane] = sc[lane] - lse;
}

// ---------------------------------------------------------------------------
// Launch
// ---------------------------------------------------------------------------
extern "C" void kernel_launch(void* const* d_in, const int* in_sizes, int n_in,
                              void* d_out, int out_size, void* d_ws,
                              size_t ws_size, hipStream_t stream) {
  (void)in_sizes; (void)n_in; (void)out_size; (void)ws_size;
  const int*           cand   = (const int*)d_in[0];
  const int*           clk    = (const int*)d_in[1];
  const unsigned char* mask   = (const unsigned char*)d_in[2];
  const float*         emb    = (const float*)d_in[3];
  const float*         w1     = (const float*)d_in[4];
  const float*         b1     = (const float*)d_in[5];
  const float*         w2     = (const float*)d_in[6];
  const float*         b2     = (const float*)d_in[7];
  const float*         w3     = (const float*)d_in[8];
  const float*         b3     = (const float*)d_in[9];
  const float*         ln_w   = (const float*)d_in[10];
  const float*         ln_b   = (const float*)d_in[11];
  const float*         q_w    = (const float*)d_in[12];
  const float*         q_l    = (const float*)d_in[13];
  const float*         c3w1   = (const float*)d_in[14];
  const float*         c3b1   = (const float*)d_in[15];
  const float*         c3w2   = (const float*)d_in[16];
  const float*         c3b2   = (const float*)d_in[17];
  const float*         ltr_w  = (const float*)d_in[18];
  const float*         ltr_b  = (const float*)d_in[19];
  float*               out    = (float*)d_out;

  char*  base = (char*)d_ws;
  size_t off  = 0;
  auto take   = [&](size_t bytes) {
    char* p = base + off;
    off     = (off + bytes + 255) & ~(size_t)255;
    return p;
  };
  _Float16* Bt1   = (_Float16*)take((size_t)FP * K1P * 2);
  _Float16* Bt2   = (_Float16*)take((size_t)FP * K23P * 2);
  _Float16* Bt3   = (_Float16*)take((size_t)FP * K23P * 2);
  char*     bigA  = take((size_t)NROW * K1P * 2);          // A1, later fusion
  float*    pre   = (float*)take((size_t)NROW * FP * 4);   // later pooled1
  _Float16* d1    = (_Float16*)take((size_t)NROW * FP * 2);
  _Float16* d2    = (_Float16*)take((size_t)NROW * FP * 2);
  _Float16* d3    = (_Float16*)take((size_t)NROW * FP * 2);
  float*    reprs = (float*)take((size_t)NSEQ * FP * 4);
  int*      idx   = (int*)take((size_t)Bx * CDD * Kx * 4);
  float*    pool2 = (float*)take((size_t)160 * 16 * 12 * 4);
  _Float16* A1    = (_Float16*)bigA;
  float*    fus   = (float*)bigA;     // reuse after GEMM1 done
  float*    pool1 = pre;              // reuse after LN3 done

  const int WAVES_PER_BLK = 8;
  const int GEMM_TILES    = (NROW / 16) * (FP / 16);  // 22000
  const int GEMM_BLKS     = (GEMM_TILES + WAVES_PER_BLK - 1) / WAVES_PER_BLK;

  // pack weights to f16
  pack_w_kernel<<<580, 256, 0, stream>>>(w1, Bt1, Ex, Ex, K1P);
  pack_w_kernel<<<300, 256, 0, stream>>>(w2, Bt2, Fx, FP, K23P);
  pack_w_kernel<<<300, 256, 0, stream>>>(w3, Bt3, Fx, FP, K23P);

  // layer 1: gather+im2col, async-LDS WMMA GEMM, LN
  im2col1_kernel<<<NROW, 128, 0, stream>>>(cand, clk, emb, A1);
  gemm1_lds_kernel<<<NROW / 16, 320, 0, stream>>>(A1, Bt1, b1, pre);
  ln_kernel<<<(NROW + 7) / 8, 256, 0, stream>>>(pre, ln_w, ln_b, d1);

  // layer 2 (dilation 2) and layer 3 (dilation 3): implicit-im2col WMMA GEMMs
  gemm_dilated_kernel<<<GEMM_BLKS, 256, 0, stream>>>(d1, Bt2, b2, pre, 2);
  ln_kernel<<<(NROW + 7) / 8, 256, 0, stream>>>(pre, ln_w, ln_b, d2);
  gemm_dilated_kernel<<<GEMM_BLKS, 256, 0, stream>>>(d2, Bt3, b3, pre, 3);
  ln_kernel<<<(NROW + 7) / 8, 256, 0, stream>>>(pre, ln_w, ln_b, d3);

  // attention -> reprs, then scores + top-K indices
  repr_kernel<<<NSEQ, 256, 0, stream>>>(d1, d2, d3, q_l, q_w, reprs);
  topk_kernel<<<Bx * CDD, 256, 0, stream>>>(reprs, mask, idx);

  // fusion einsum (WMMA), then fused conv3d+relu+pool stages, final head
  fusion_kernel<<<(NUNIT + WAVES_PER_BLK - 1) / WAVES_PER_BLK, 256, 0, stream>>>(
      d1, d2, d3, idx, fus);
  conv3d1_kernel<<<7200, 256, 0, stream>>>(fus, c3w1, c3b1, pool1);
  conv3d2_kernel<<<120, 256, 0, stream>>>(pool1, c3w2, c3b2, pool2);
  final_kernel<<<Bx, 32, 0, stream>>>(pool2, ltr_w, ltr_b, out);
}
